// SingleStreamBlock_35785667510390
// MI455X (gfx1250) — matrix operand order repaired and checked by reading, not compiled, not run on hardware
//
#include <hip/hip_runtime.h>
#include <hip/hip_bf16.h>

// ---------------- problem constants ----------------
#define Hn    24
#define DHn   128
#define Dn    3072
#define MLPn  12288
#define LTXT  512
#define LIMG  1024
#define Ln    1536           // LTXT + LIMG
#define NQKV  9216           // 3*D
#define N1    21504          // 3*D + MLP
#define NCAT  15360          // D + MLP

typedef __attribute__((ext_vector_type(16))) __bf16 v16bf;
typedef __attribute__((ext_vector_type(8)))  __bf16 v8bf;
typedef __attribute__((ext_vector_type(8)))  float  v8f;

union BF16x16 { v16bf v; v8bf h[2]; };

// ---- Tensor Data Mover availability (toolchain-dependent arity) ----
#if defined(__gfx1250__) && __has_builtin(__builtin_amdgcn_tensor_load_to_lds) && \
    __has_builtin(__builtin_amdgcn_s_wait_tensorcnt)
#define HAVE_TDM 1
#if __has_include(<hip/amd_detail/amd_gfx1250_TDM.h>)
#define TDM_6ARG 1
#endif
#endif

typedef __attribute__((ext_vector_type(4))) unsigned int u32x4;
typedef __attribute__((ext_vector_type(4))) int          i32x4;
typedef __attribute__((ext_vector_type(8))) int          i32x8;

#if HAVE_TDM
// 2D bf16 tile load via TDM with LDS row padding done by the descriptor.
// Descriptor built per CDNA5 ISA D# Group0/Group1 bit layout:
//   G0: count=1 | lds_addr | global_addr(57b) | type=2
//   G1: data_size=1(2B) | pad_enable | pad_interval | pad_amount |
//       tensor_dim0/1 | tile_dim0/1 | tensor_dim0_stride
__device__ __forceinline__ void tdm_load_tile_bf16(
    unsigned lds_byte_addr, const unsigned short* gsrc,
    unsigned tile_d0, unsigned tile_d1,
    unsigned tensor_d0, unsigned tensor_d1, unsigned stride_d0,
    unsigned pad_interval, unsigned pad_amount) {
  unsigned long long ga = (unsigned long long)(size_t)gsrc;
  u32x4 g0;
  g0[0] = 1u;                                        // count = 1 valid descriptor
  g0[1] = lds_byte_addr;                             // lds_addr
  g0[2] = (unsigned)(ga & 0xffffffffu);              // global_addr[31:0]
  g0[3] = (unsigned)((ga >> 32) & 0x01ffffffu)       // global_addr[56:32]
          | 0x80000000u;                             // type = 2 ("image")
  i32x8 g1;
  g1[0] = (int)((1u << 16)                           // data_size = 2 bytes
                | (1u << 20)                         // pad_enable
                | (pad_interval << 22)
                | (pad_amount << 25));
  g1[1] = (int)((tensor_d0 & 0xffffu) << 16);        // tensor_dim0[15:0]
  g1[2] = (int)(((tensor_d0 >> 16) & 0xffffu)        // tensor_dim0[31:16]
                | ((tensor_d1 & 0xffffu) << 16));    // tensor_dim1[15:0]
  g1[3] = (int)(((tensor_d1 >> 16) & 0xffffu)        // tensor_dim1[31:16]
                | (tile_d0 << 16));                  // tile_dim0
  g1[4] = (int)(tile_d1 & 0xffffu);                  // tile_dim1 (tile_dim2 = 0)
  g1[5] = (int)stride_d0;                            // tensor_dim0_stride[31:0]
  g1[6] = 0;                                         // stride hi / dim1_stride lo
  g1[7] = 0;
  i32x4 gz = {0, 0, 0, 0};
#if TDM_6ARG
  i32x8 gz8 = {0, 0, 0, 0, 0, 0, 0, 0};
  __builtin_amdgcn_tensor_load_to_lds(g0, g1, gz, gz, gz8, 0);
#else
  __builtin_amdgcn_tensor_load_to_lds(g0, g1, gz, gz, 0);
#endif
}
#endif  // HAVE_TDM

__device__ __forceinline__ unsigned short f2bf(float f) {
  union { float f; unsigned int u; } c; c.f = f;
  unsigned int u = c.u;
  u += 0x7FFFu + ((u >> 16) & 1u);   // round-to-nearest-even
  return (unsigned short)(u >> 16);
}

__device__ __forceinline__ float gelu_tanh(float x) {
  float x3 = x * x * x;
  return 0.5f * x * (1.0f + tanhf(0.7978845608028654f * (x + 0.044715f * x3)));
}

// ---------------- kernel 1a: silu(vec) ----------------
__global__ __launch_bounds__(256) void silu_kernel(const float* __restrict__ vec,
                                                   float* __restrict__ sv) {
  int i = blockIdx.x * 256 + threadIdx.x;
  if (i < Dn) {
    float v = vec[i];
    sv[i] = v / (1.0f + __expf(-v));
  }
}

// ---------------- kernel 1b: mod = silu(vec) @ mod_w^T + mod_b ----------------
__global__ __launch_bounds__(256) void modgemv_kernel(const float* __restrict__ sv,
                                                      const float* __restrict__ mod_w,
                                                      const float* __restrict__ mod_b,
                                                      float* __restrict__ modv) {
  int lane = threadIdx.x & 31;
  int wid  = threadIdx.x >> 5;
  for (int q = 0; q < 4; ++q) {
    int j = blockIdx.x * 32 + wid * 4 + q;   // output index, < 9216
    const float* wr = mod_w + (size_t)j * Dn;
    float acc = 0.0f;
    for (int it = 0; it < Dn / 128; ++it) {
      int k = lane * 4 + it * 128;
      float4 w = *(const float4*)(wr + k);
      float4 s = *(const float4*)(sv + k);
      acc += w.x * s.x + w.y * s.y + w.z * s.z + w.w * s.w;
    }
    for (int m = 1; m < 32; m <<= 1) acc += __shfl_xor(acc, m, 32);
    if (lane == 0) modv[j] = acc + mod_b[j];
  }
}

// ---------------- kernel 2: LayerNorm + modulation -> x_mod (bf16) ----------------
__global__ __launch_bounds__(256) void ln_mod_kernel(const float* __restrict__ txt,
                                                     const float* __restrict__ img,
                                                     const float* __restrict__ modv,
                                                     unsigned short* __restrict__ xmod) {
  __shared__ float red[8];
  int row = blockIdx.x;
  int tid = threadIdx.x, lane = tid & 31, wid = tid >> 5;
  const float* src = (row < LTXT) ? (txt + (size_t)row * Dn)
                                  : (img + (size_t)(row - LTXT) * Dn);
  float vals[12];
  float s = 0.0f;
  #pragma unroll
  for (int i = 0; i < 12; ++i) { vals[i] = src[tid + i * 256]; s += vals[i]; }
  for (int m = 1; m < 32; m <<= 1) s += __shfl_xor(s, m, 32);
  if (lane == 0) red[wid] = s;
  __syncthreads();
  float tot = 0.0f;
  #pragma unroll
  for (int w = 0; w < 8; ++w) tot += red[w];
  float mu = tot * (1.0f / Dn);
  __syncthreads();
  float vs = 0.0f;
  #pragma unroll
  for (int i = 0; i < 12; ++i) { float d = vals[i] - mu; vs += d * d; }
  for (int m = 1; m < 32; m <<= 1) vs += __shfl_xor(vs, m, 32);
  if (lane == 0) red[wid] = vs;
  __syncthreads();
  float vtot = 0.0f;
  #pragma unroll
  for (int w = 0; w < 8; ++w) vtot += red[w];
  float rstd = rsqrtf(vtot * (1.0f / Dn) + 1e-6f);
  #pragma unroll
  for (int i = 0; i < 12; ++i) {
    int n = tid + i * 256;
    float ln = (vals[i] - mu) * rstd;
    float out = (1.0f + modv[Dn + n]) * ln + modv[n];
    xmod[(size_t)row * Dn + n] = f2bf(out);
  }
}

// ---------------- WMMA GEMM: C[M,N] = A(bf16)[M,K] * W(f32,row-major [N,K])^T ----------------
// Double-buffered LDS pipeline: TDM moves the bf16 A tile for iteration i+1
// while the WMMAs for iteration i run; the fp32 W tile for i+1 is staged in
// VGPRs before the WMMAs and converted/stored to LDS after them, so the
// global-load wait lands behind the matrix work. One barrier per iteration.
// mode 0: GEMM1 epilogue (qkv fp32 / gelu->cat bf16)
// mode 1: GEMM2 epilogue (bias, gate, residual, scatter to d_out)
__global__ __launch_bounds__(256) void wmma_gemm_kernel(
    const unsigned short* __restrict__ A, const float* __restrict__ W,
    const float* __restrict__ bias, int K, int mode,
    float* __restrict__ qkv_out, unsigned short* __restrict__ cat,
    const float* __restrict__ txt, const float* __restrict__ img,
    const float* __restrict__ modv, float* __restrict__ dout) {
  __shared__ unsigned short As[2][128 * 40];   // 128 rows x 32 k, 80B padded rows
  __shared__ unsigned short Ws[2][128 * 40];
  int tid = threadIdx.x, lane = tid & 31, wid = tid >> 5;
  int l15 = lane & 15, lh = lane >> 4;
  int mblk = blockIdx.y, nblk = blockIdx.x;
  int waveM = wid >> 2, waveN = wid & 3;

  v8f acc[4][2];
  #pragma unroll
  for (int mi = 0; mi < 4; ++mi)
    #pragma unroll
    for (int ni = 0; ni < 2; ++ni)
      #pragma unroll
      for (int r = 0; r < 8; ++r) acc[mi][ni][r] = 0.0f;

  float4 wreg[4];
#if !HAVE_TDM
  uint4 areg[2];
#endif

  auto issueA = [&](int buf, int k0) {
#if HAVE_TDM
    if (wid == 0) {
      tdm_load_tile_bf16((unsigned)(size_t)(void*)&As[buf][0],
                         A + (size_t)(mblk * 128) * K + k0,
                         /*tile_d0=*/32, /*tile_d1=*/128,
                         /*tensor_d0=*/(unsigned)K, /*tensor_d1=*/Ln,
                         /*stride_d0=*/(unsigned)K,
                         /*pad_interval=*/3u, /*pad_amount=*/3u);
    }
#else
    #pragma unroll
    for (int i = 0; i < 2; ++i) {
      int c = tid + i * 256;
      int r = c >> 2, sgl = c & 3;
      areg[i] = *(const uint4*)(A + (size_t)(mblk * 128 + r) * K + k0 + sgl * 8);
    }
#endif
  };
  auto commitA = [&](int buf) {
#if !HAVE_TDM
    #pragma unroll
    for (int i = 0; i < 2; ++i) {
      int c = tid + i * 256;
      int r = c >> 2, sgl = c & 3;
      *(uint4*)&As[buf][r * 40 + sgl * 8] = areg[i];
    }
#endif
  };
  auto issueW = [&](int k0) {
    #pragma unroll
    for (int i = 0; i < 4; ++i) {
      int c = tid + i * 256;
      int r = c >> 3, sgl = c & 7;
      const float* wp = W + (size_t)(nblk * 128 + r) * K + k0 + sgl * 4;
      wreg[i] = *(const float4*)wp;
      if (k0 + 64 < K) __builtin_prefetch(wp + 64, 0, 2);  // 2 K-tiles ahead
    }
  };
  auto commitW = [&](int buf) {
    #pragma unroll
    for (int i = 0; i < 4; ++i) {
      int c = tid + i * 256;
      int r = c >> 3, sgl = c & 7;
      uint2 p;
      p.x = (unsigned)f2bf(wreg[i].x) | ((unsigned)f2bf(wreg[i].y) << 16);
      p.y = (unsigned)f2bf(wreg[i].z) | ((unsigned)f2bf(wreg[i].w) << 16);
      *(uint2*)&Ws[buf][r * 40 + sgl * 4] = p;
    }
  };

  // prologue: fill buffer 0
  issueA(0, 0);
  issueW(0);
  commitA(0);
  commitW(0);
#if HAVE_TDM
  if (wid == 0) __builtin_amdgcn_s_wait_tensorcnt(0);
#endif
  __syncthreads();

  int nIter = K / 32;
  for (int it = 0; it < nIter; ++it) {
    int cur = it & 1, nxt = cur ^ 1;
    bool more = (it + 1) < nIter;
    if (more) {               // start fetching tile i+1 (TDM + global loads)
      issueA(nxt, (it + 1) * 32);
      issueW((it + 1) * 32);
    }

    // compute tile i
    BF16x16 bfr[2];
    int kh = lh * 8;
    #pragma unroll
    for (int ni = 0; ni < 2; ++ni) {
      int n = waveN * 32 + ni * 16 + l15;
      bfr[ni].h[0] = *(const v8bf*)&Ws[cur][n * 40 + kh];
      bfr[ni].h[1] = *(const v8bf*)&Ws[cur][n * 40 + kh + 16];
    }
    int kb = lh * 8;
    #pragma unroll
    for (int mi = 0; mi < 4; ++mi) {
      int r = waveM * 64 + mi * 16 + l15;
      BF16x16 a;
      a.h[0] = *(const v8bf*)&As[cur][r * 40 + kb];
      a.h[1] = *(const v8bf*)&As[cur][r * 40 + 16 + kb];
      #pragma unroll
      for (int ni = 0; ni < 2; ++ni)
        acc[mi][ni] = __builtin_amdgcn_wmma_f32_16x16x32_bf16(
            false, a.v, false, bfr[ni].v, (short)0, acc[mi][ni], false, false);
    }

    if (more) {               // publish tile i+1 (loadcnt wait lands here)
      commitA(nxt);
      commitW(nxt);
#if HAVE_TDM
      if (wid == 0) __builtin_amdgcn_s_wait_tensorcnt(0);
#endif
    }
    __syncthreads();
  }

  // epilogue
  #pragma unroll
  for (int mi = 0; mi < 4; ++mi) {
    #pragma unroll
    for (int ni = 0; ni < 2; ++ni) {
      int rowb = mblk * 128 + waveM * 64 + mi * 16 + lh * 8;
      int col  = nblk * 128 + waveN * 32 + ni * 16 + l15;
      float b = bias[col];
      #pragma unroll
      for (int r = 0; r < 8; ++r) {
        int m = rowb + r;
        float val = acc[mi][ni][r] + b;
        if (mode == 0) {
          if (col < NQKV) {
            qkv_out[(size_t)m * NQKV + col] = val;
          } else {
            cat[(size_t)m * NCAT + Dn + (col - NQKV)] = f2bf(gelu_tanh(val));
          }
        } else {
          float xv = (m < LTXT) ? txt[(size_t)m * Dn + col]
                                : img[(size_t)(m - LTXT) * Dn + col];
          float res = xv + modv[2 * Dn + col] * val;
          if (m < LTXT) dout[(size_t)LIMG * Dn + (size_t)m * Dn + col] = res;
          else          dout[(size_t)(m - LTXT) * Dn + col] = res;
        }
      }
    }
  }
}

// ---------------- kernel 4: RMSNorm(q,k) + RoPE -> q,k,v bf16 [H][L][128] ----------------
__global__ __launch_bounds__(256) void rope_rms_kernel(
    const float* __restrict__ qkv, const float* __restrict__ q_scale,
    const float* __restrict__ k_scale, const float* __restrict__ txt_pe,
    const float* __restrict__ img_pe, unsigned short* __restrict__ qO,
    unsigned short* __restrict__ kO, unsigned short* __restrict__ vO) {
  int l = blockIdx.x;
  int lane = threadIdx.x & 31, wid = threadIdx.x >> 5;
  const float* pe = (l < LTXT) ? (txt_pe + (size_t)l * 256)
                               : (img_pe + (size_t)(l - LTXT) * 256);
  #pragma unroll
  for (int j = 0; j < 3; ++j) {
    int h = wid + 8 * j;
    size_t base = (size_t)l * NQKV + h * DHn + lane * 4;
    float4 qv = *(const float4*)(qkv + base);
    float4 kv = *(const float4*)(qkv + base + Dn);
    float4 vv = *(const float4*)(qkv + base + 2 * Dn);
    float sq = qv.x * qv.x + qv.y * qv.y + qv.z * qv.z + qv.w * qv.w;
    float sk = kv.x * kv.x + kv.y * kv.y + kv.z * kv.z + kv.w * kv.w;
    for (int m = 1; m < 32; m <<= 1) {
      sq += __shfl_xor(sq, m, 32);
      sk += __shfl_xor(sk, m, 32);
    }
    float rq = rsqrtf(sq * (1.0f / DHn) + 1e-6f);
    float rk = rsqrtf(sk * (1.0f / DHn) + 1e-6f);
    float qs[4], ks[4];
    float qf[4] = {qv.x, qv.y, qv.z, qv.w};
    float kf[4] = {kv.x, kv.y, kv.z, kv.w};
    #pragma unroll
    for (int i = 0; i < 4; ++i) {
      qs[i] = qf[i] * rq * q_scale[lane * 4 + i];
      ks[i] = kf[i] * rk * k_scale[lane * 4 + i];
    }
    float qo[4], ko[4];
    #pragma unroll
    for (int p = 0; p < 2; ++p) {
      int pi = lane * 2 + p;
      const float* pb = pe + pi * 4;   // pb[0]=r0c0 pb[1]=r0c1 pb[2]=r1c0 pb[3]=r1c1
      float e0 = qs[2 * p], e1 = qs[2 * p + 1];
      qo[2 * p]     = pb[0] * e0 + pb[1] * e1;
      qo[2 * p + 1] = pb[2] * e0 + pb[3] * e1;
      float f0 = ks[2 * p], f1 = ks[2 * p + 1];
      ko[2 * p]     = pb[0] * f0 + pb[1] * f1;
      ko[2 * p + 1] = pb[2] * f0 + pb[3] * f1;
    }
    size_t ob = ((size_t)h * Ln + l) * DHn + lane * 4;
    ushort4 oq, ok, ov;
    oq.x = f2bf(qo[0]); oq.y = f2bf(qo[1]); oq.z = f2bf(qo[2]); oq.w = f2bf(qo[3]);
    ok.x = f2bf(ko[0]); ok.y = f2bf(ko[1]); ok.z = f2bf(ko[2]); ok.w = f2bf(ko[3]);
    ov.x = f2bf(vv.x);  ov.y = f2bf(vv.y);  ov.z = f2bf(vv.z);  ov.w = f2bf(vv.w);
    *(ushort4*)(qO + ob) = oq;
    *(ushort4*)(kO + ob) = ok;
    *(ushort4*)(vO + ob) = ov;
  }
}

// ---------------- kernel 5: flash attention (WMMA) ----------------
__global__ __launch_bounds__(256) void attn_kernel(
    const unsigned short* __restrict__ Q, const unsigned short* __restrict__ Kb,
    const unsigned short* __restrict__ Vb, unsigned short* __restrict__ cat) {
  __shared__ unsigned short Ks[32 * 136];   // 32 keys x 128 dims, rows padded to 272B
  __shared__ unsigned short Vt[128 * 40];   // transposed: 128 dims x 32 keys, rows 80B
  __shared__ unsigned short Ps[8 * 16 * 40];// per-wave P tile, 16 x 32 (rows 80B)
  int tid = threadIdx.x, lane = tid & 31, wid = tid >> 5;
  int l15 = lane & 15, lh = lane >> 4;
  int h = blockIdx.y, mblk = blockIdx.x;
  int qbase = mblk * 128 + wid * 16;

  // preload Q fragments (16 rows x 128 dims -> 4 K-steps) in native A layout
  BF16x16 aq[4];
  {
    const unsigned short* qp = Q + ((size_t)h * Ln + qbase + l15) * DHn;
    int kb = lh * 8;
    #pragma unroll
    for (int ks = 0; ks < 4; ++ks) {
      aq[ks].h[0] = *(const v8bf*)(qp + ks * 32 + kb);
      aq[ks].h[1] = *(const v8bf*)(qp + ks * 32 + 16 + kb);
    }
  }

  v8f O[8];
  float mrow[8], lsum[8];
  #pragma unroll
  for (int t = 0; t < 8; ++t) {
    #pragma unroll
    for (int r = 0; r < 8; ++r) O[t][r] = 0.0f;
  }
  #pragma unroll
  for (int r = 0; r < 8; ++r) { mrow[r] = -1e30f; lsum[r] = 0.0f; }

  unsigned short* myPs = Ps + wid * 16 * 40;
  const float sc = 0.08838834764831845f;    // 1/sqrt(128)

  for (int kt = 0; kt < Ln / 32; ++kt) {
    __syncthreads();
#if HAVE_TDM
    // K tile straight copy via TDM (64DW data + 4DW pad -> 272B rows)
    if (wid == 0) {
      tdm_load_tile_bf16((unsigned)(size_t)(void*)&Ks[0],
                         Kb + ((size_t)h * Ln + kt * 32) * DHn,
                         /*tile_d0=*/128, /*tile_d1=*/32,
                         /*tensor_d0=*/128u, /*tensor_d1=*/Ln,
                         /*stride_d0=*/128u,
                         /*pad_interval=*/5u, /*pad_amount=*/3u);
    }
#endif
    // V tile transposed into LDS (manual; TDM cannot transpose)
    #pragma unroll
    for (int i = 0; i < 2; ++i) {
      int c = tid + i * 256;
      int kr = c >> 4, sg = c & 15;
#if !HAVE_TDM
      const unsigned short* kp = Kb + ((size_t)h * Ln + kt * 32 + kr) * DHn + sg * 8;
      *(uint4*)&Ks[kr * 136 + sg * 8] = *(const uint4*)kp;
#endif
      uint4 vv = *(const uint4*)(Vb + ((size_t)h * Ln + kt * 32 + kr) * DHn + sg * 8);
      const unsigned short* vs = (const unsigned short*)&vv;
      #pragma unroll
      for (int jj = 0; jj < 8; ++jj) Vt[(sg * 8 + jj) * 40 + kr] = vs[jj];
    }
#if HAVE_TDM
    if (wid == 0) __builtin_amdgcn_s_wait_tensorcnt(0);
#endif
    __syncthreads();

    // S = Q * K^T for 32 keys (two 16-key subtiles)
    v8f s0, s1;
    #pragma unroll
    for (int r = 0; r < 8; ++r) { s0[r] = 0.0f; s1[r] = 0.0f; }
    int kh = lh * 8;
    #pragma unroll
    for (int ks = 0; ks < 4; ++ks) {
      BF16x16 b0, b1;
      b0.h[0] = *(const v8bf*)&Ks[l15 * 136 + ks * 32 + kh];
      b0.h[1] = *(const v8bf*)&Ks[l15 * 136 + ks * 32 + kh + 16];
      b1.h[0] = *(const v8bf*)&Ks[(16 + l15) * 136 + ks * 32 + kh];
      b1.h[1] = *(const v8bf*)&Ks[(16 + l15) * 136 + ks * 32 + kh + 16];
      s0 = __builtin_amdgcn_wmma_f32_16x16x32_bf16(false, aq[ks].v, false, b0.v,
                                                   (short)0, s0, false, false);
      s1 = __builtin_amdgcn_wmma_f32_16x16x32_bf16(false, aq[ks].v, false, b1.v,
                                                   (short)0, s1, false, false);
    }

    // online softmax (rows live across 16-lane halves)
    #pragma unroll
    for (int r = 0; r < 8; ++r) {
      float a = s0[r] * sc, b = s1[r] * sc;
      float v = fmaxf(a, b);
      for (int m = 1; m < 16; m <<= 1) v = fmaxf(v, __shfl_xor(v, m, 32));
      float nm = fmaxf(mrow[r], v);
      float p0 = __expf(a - nm), p1 = __expf(b - nm);
      float rs = p0 + p1;
      for (int m = 1; m < 16; m <<= 1) rs += __shfl_xor(rs, m, 32);
      float scl = __expf(mrow[r] - nm);
      lsum[r] = lsum[r] * scl + rs;
      mrow[r] = nm;
      #pragma unroll
      for (int t = 0; t < 8; ++t) O[t][r] *= scl;
      int rl = r + 8 * lh;
      myPs[rl * 40 + l15]      = f2bf(p0);
      myPs[rl * 40 + 16 + l15] = f2bf(p1);
    }

    // P (16x32) back into A layout, then O += P * V
    BF16x16 pa;
    int kb = lh * 8;
    pa.h[0] = *(const v8bf*)&myPs[l15 * 40 + kb];
    pa.h[1] = *(const v8bf*)&myPs[l15 * 40 + 16 + kb];
    #pragma unroll
    for (int t = 0; t < 8; ++t) {
      BF16x16 vb2;
      int n = t * 16 + l15;
      vb2.h[0] = *(const v8bf*)&Vt[n * 40 + kh];
      vb2.h[1] = *(const v8bf*)&Vt[n * 40 + kh + 16];
      O[t] = __builtin_amdgcn_wmma_f32_16x16x32_bf16(false, pa.v, false, vb2.v,
                                                     (short)0, O[t], false, false);
    }
  }

  // normalize and write attn into concat buffer cols [0, 3072)
  #pragma unroll
  for (int r = 0; r < 8; ++r) {
    float inv = 1.0f / lsum[r];
    int tok = qbase + r + 8 * lh;
    #pragma unroll
    for (int t = 0; t < 8; ++t) {
      cat[(size_t)tok * NCAT + h * DHn + t * 16 + l15] = f2bf(O[t][r] * inv);
    }
  }
}

// ---------------- host launcher ----------------
extern "C" void kernel_launch(void* const* d_in, const int* in_sizes, int n_in,
                              void* d_out, int out_size, void* d_ws, size_t ws_size,
                              hipStream_t stream) {
  const float* img     = (const float*)d_in[0];
  const float* txt     = (const float*)d_in[1];
  const float* vec     = (const float*)d_in[2];
  const float* txt_pe  = (const float*)d_in[3];
  const float* img_pe  = (const float*)d_in[4];
  const float* lin1_w  = (const float*)d_in[5];
  const float* lin1_b  = (const float*)d_in[6];
  const float* lin2_w  = (const float*)d_in[7];
  const float* lin2_b  = (const float*)d_in[8];
  const float* mod_w   = (const float*)d_in[9];
  const float* mod_b   = (const float*)d_in[10];
  const float* q_scale = (const float*)d_in[11];
  const float* k_scale = (const float*)d_in[12];
  float* dout = (float*)d_out;
  char*  ws   = (char*)d_ws;

  size_t off = 0;
  auto carve = [&](size_t bytes) { size_t r = off; off = (off + bytes + 255) & ~(size_t)255; return r; };
  float*          sv   = (float*)(ws + carve((size_t)Dn * 4));
  float*          modv = (float*)(ws + carve((size_t)NQKV * 4));
  unsigned short* xmod = (unsigned short*)(ws + carve((size_t)Ln * Dn * 2));
  float*          qkv  = (float*)(ws + carve((size_t)Ln * NQKV * 4));
  unsigned short* cat  = (unsigned short*)(ws + carve((size_t)Ln * NCAT * 2));
  unsigned short* qb   = (unsigned short*)(ws + carve((size_t)Hn * Ln * DHn * 2));
  unsigned short* kb   = (unsigned short*)(ws + carve((size_t)Hn * Ln * DHn * 2));
  unsigned short* vb   = (unsigned short*)(ws + carve((size_t)Hn * Ln * DHn * 2));

  silu_kernel<<<Dn / 256, 256, 0, stream>>>(vec, sv);
  modgemv_kernel<<<NQKV / 32, 256, 0, stream>>>(sv, mod_w, mod_b, modv);
  ln_mod_kernel<<<Ln, 256, 0, stream>>>(txt, img, modv, xmod);

  // GEMM1: y = x_mod @ lin1_w^T + b ; qkv -> fp32 scratch, mlp -> gelu -> cat (bf16)
  wmma_gemm_kernel<<<dim3(N1 / 128, Ln / 128), 256, 0, stream>>>(
      xmod, lin1_w, lin1_b, Dn, 0, qkv, cat, nullptr, nullptr, nullptr, nullptr);

  rope_rms_kernel<<<Ln, 256, 0, stream>>>(qkv, q_scale, k_scale, txt_pe, img_pe,
                                          qb, kb, vb);

  attn_kernel<<<dim3(Ln / 128, Hn), 256, 0, stream>>>(qb, kb, vb, cat);

  // GEMM2: out = cat @ lin2_w^T + b ; epilogue: x + gate*out, scatter (img, txt)
  wmma_gemm_kernel<<<dim3(Dn / 128, Ln / 128), 256, 0, stream>>>(
      cat, lin2_w, lin2_b, NCAT, 1, nullptr, nullptr, txt, img, modv, dout);
}
